// TERT_85177791414517
// MI455X (gfx1250) — matrix operand order repaired and checked by reading, not compiled, not run on hardware
//
#include <hip/hip_runtime.h>
#include <hip/hip_bf16.h>
#include <math.h>

// ---------------------------------------------------------------------------
// MI455X (gfx1250) implementation: persistent mixed-precision LSTM with
// v_wmma_f32_16x16x32_f16, full weight residency in 320KB LDS, f32 state in
// VGPRs, f16 only on the matmul path. FSim applied only to gathered rows.
// ---------------------------------------------------------------------------

#define BATCH 256
#define TLEN  1024
#define HDIM  128
#define G4    512            // 4*H gates
#define NROWS_G 11008        // 3*256 (hT) + 4*2560 (gathered y rows)
#define TA_OFF  768
#define TF_OFF  3328
#define A_OFF   5888
#define F_OFF   8448

typedef _Float16 v16h __attribute__((ext_vector_type(16)));
typedef _Float16 v8h  __attribute__((ext_vector_type(8)));
typedef float    v8f  __attribute__((ext_vector_type(8)));

__device__ __forceinline__ float sigf(float x) { return 1.0f / (1.0f + __expf(-x)); }

// A fragment (16x32 f16, row-major [16][128] source in LDS).
// lanes 0-15: M=lane, halves 0..15 = K+0..15
// lanes 16-31: M=lane-16, halves 0..15 = K+16..31
__device__ __forceinline__ v16h load_a_frag(const _Float16* __restrict__ src, int lane, int kk) {
  int row = lane & 15;
  int k0  = kk * 32 + ((lane & 16) ? 16 : 0);
  const _Float16* p = src + row * HDIM + k0;
  v8h lo = *(const v8h*)p;
  v8h hi = *(const v8h*)(p + 8);
  v16h a;
#pragma unroll
  for (int i = 0; i < 8; ++i) { a[i] = lo[i]; a[i + 8] = hi[i]; }
  return a;
}

// B fragment (32x16 f16) from W stored row-major [N][128] (i.e. W^T columns).
// lanes 0-15: N=ncol, halves = K+0..15 ; lanes 16-31: halves = K+16..31
__device__ __forceinline__ v16h load_b_frag(const _Float16* __restrict__ W, int ncol, int lane, int kk) {
  int k0 = kk * 32 + ((lane & 16) ? 16 : 0);
  return *(const v16h*)(W + ncol * HDIM + k0);
}

__device__ __forceinline__ v8f wmma16(v16h a, v16h b, v8f c) {
  return __builtin_amdgcn_wmma_f32_16x16x32_f16(false, a, false, b, (short)0, c, false, false);
}

// ---------------------------------------------------------------------------
// Weight preparation: f32 -> f16 copies laid out for B-fragment loads.
// ---------------------------------------------------------------------------
__global__ void prep_weights(const float* __restrict__ W_ih, const float* __restrict__ W_hh,
                             const float* __restrict__ b_ih, const float* __restrict__ b_hh,
                             const float* __restrict__ Wi, const float* __restrict__ Wc,
                             const float* __restrict__ Wo,
                             _Float16* __restrict__ gWih, _Float16* __restrict__ gWhh,
                             float* __restrict__ gWgps, float* __restrict__ gBias,
                             _Float16* __restrict__ gWi, _Float16* __restrict__ gWc,
                             _Float16* __restrict__ gWo) {
  int i = blockIdx.x * blockDim.x + threadIdx.x;
  if (i < G4 * HDIM) {
    int n = i >> 7, k = i & 127;
    gWih[i] = (_Float16)W_ih[n * 130 + k];   // embedding part only (K=0..127)
    gWhh[i] = (_Float16)W_hh[i];
  }
  if (i < G4 * 2) { int n = i >> 1, c = i & 1; gWgps[i] = W_ih[n * 130 + 128 + c]; }
  if (i < G4) gBias[i] = b_ih[i] + b_hh[i];
  if (i < HDIM * HDIM) {
    gWi[i] = (_Float16)Wi[i]; gWc[i] = (_Float16)Wc[i]; gWo[i] = (_Float16)Wo[i];
  }
}

// ---------------------------------------------------------------------------
// Persistent masked LSTM. 16 workgroups x 16 batch rows. Both weight matrices
// resident in LDS (f16). Wave w owns hidden columns [16w,16w+16) and computes
// its own i/f/g/o tiles -> nonlinearity is wave-local; c,h kept in f32 VGPRs.
// ---------------------------------------------------------------------------
__global__ __launch_bounds__(256) void lstm_kernel(
    const float* __restrict__ gps, const int* __restrict__ grid_idx,
    const int* __restrict__ length, const float* __restrict__ E,
    const _Float16* __restrict__ gWih, const _Float16* __restrict__ gWhh,
    const float* __restrict__ gWgps, const float* __restrict__ gBias,
    float* __restrict__ y, float* __restrict__ hT) {
  extern __shared__ char smem[];
  _Float16* sWih = (_Float16*)smem;            // 512*128 f16
  _Float16* sWhh = sWih + G4 * HDIM;           // 512*128 f16
  float* sWg = (float*)(sWhh + G4 * HDIM);     // 512*2
  float* sB  = sWg + G4 * 2;                   // 512
  _Float16* sX = (_Float16*)(sB + G4);         // 16*128 f16 (x_t embedding)
  _Float16* sH = sX + 16 * HDIM;               // 16*128 f16 (h_{t-1})
  float* sGps = (float*)(sH + 16 * HDIM);      // 16*2
  int* sLen = (int*)(sGps + 32);               // 16

  const int tid  = threadIdx.x;
  const int lane = tid & 31;
  const int wave = tid >> 5;
  const int b0   = blockIdx.x * 16;

  for (int i = tid; i < G4 * HDIM; i += 256) { sWih[i] = gWih[i]; sWhh[i] = gWhh[i]; }
  for (int i = tid; i < G4 * 2; i += 256) sWg[i] = gWgps[i];
  for (int i = tid; i < G4; i += 256) sB[i] = gBias[i];
  for (int i = tid; i < 16 * HDIM; i += 256) sH[i] = (_Float16)0.0f;
  if (tid < 16) sLen[tid] = length[b0 + tid];
  __syncthreads();

  const int mrBase = (lane & 16) ? 8 : 0;      // C/D tile: M = r + mrBase
  const int hcol   = wave * 16 + (lane & 15);  // hidden column this lane owns

  // time-invariant per-lane constants: bias + gps weight columns per gate
  float bias_g[4], wg0[4], wg1[4];
#pragma unroll
  for (int g = 0; g < 4; ++g) {
    int n = g * HDIM + hcol;
    bias_g[g] = sB[n]; wg0[g] = sWg[2 * n]; wg1[g] = sWg[2 * n + 1];
  }

  float creg[8], hreg[8];
#pragma unroll
  for (int r = 0; r < 8; ++r) { creg[r] = 0.f; hreg[r] = 0.f; }

  const int xm = tid >> 4;          // row for cooperative embedding gather
  const int xc = (tid & 15) << 3;   // 8-element chunk

  for (int t = 0; t < TLEN; ++t) {
    { // gather E[grid[b,t]] -> sX (f16), padding_idx 0 -> zeros
      int gi = grid_idx[(b0 + xm) * TLEN + t];
      v8h hx;
      if (gi == 0) {
#pragma unroll
        for (int i = 0; i < 8; ++i) hx[i] = (_Float16)0.0f;
      } else {
        const float* er = E + (size_t)gi * HDIM + xc;
#pragma unroll
        for (int i = 0; i < 8; ++i) hx[i] = (_Float16)er[i];
      }
      *(v8h*)(sX + xm * HDIM + xc) = hx;
    }
    if (tid < 32) sGps[tid] = gps[((size_t)(b0 + (tid >> 1)) * TLEN + t) * 2 + (tid & 1)];
    __syncthreads();

    // init accumulators with bias + gps contribution (K=128,129 of W_ih)
    v8f acc[4];
#pragma unroll
    for (int g = 0; g < 4; ++g) {
#pragma unroll
      for (int r = 0; r < 8; ++r) {
        int M = mrBase + r;
        acc[g][r] = bias_g[g] + sGps[2 * M] * wg0[g] + sGps[2 * M + 1] * wg1[g];
      }
    }
    // gates += x_emb @ W_ih_emb^T + h @ W_hh^T   (32 WMMAs / wave / step)
#pragma unroll
    for (int kk = 0; kk < 4; ++kk) {
      v16h ax = load_a_frag(sX, lane, kk);
      v16h ah = load_a_frag(sH, lane, kk);
#pragma unroll
      for (int g = 0; g < 4; ++g) {
        int n = g * HDIM + hcol;
        acc[g] = wmma16(ax, load_b_frag(sWih, n, lane, kk), acc[g]);
        acc[g] = wmma16(ah, load_b_frag(sWhh, n, lane, kk), acc[g]);
      }
    }
    __syncthreads();   // all sH reads complete before overwrite

#pragma unroll
    for (int r = 0; r < 8; ++r) {
      int M = mrBase + r;
      float iv = sigf(acc[0][r]);
      float fv = sigf(acc[1][r]);
      float gv = tanhf(acc[2][r]);
      float ov = sigf(acc[3][r]);
      float cn = fv * creg[r] + iv * gv;
      float hn = ov * tanhf(cn);
      float yv = 0.0f;
      if (t < sLen[M]) { creg[r] = cn; hreg[r] = hn; yv = hn; }
      y[((size_t)(b0 + M) * TLEN + t) * HDIM + hcol] = yv;
      sH[M * HDIM + hcol] = (_Float16)hreg[r];
    }
    // next iteration's first barrier covers the sH write->read hazard
  }
  __syncthreads();
#pragma unroll
  for (int r = 0; r < 8; ++r) {
    int M = mrBase + r;
    hT[(size_t)(b0 + M) * HDIM + hcol] = hreg[r];   // exact f32 state
  }
}

// ---------------------------------------------------------------------------
// Row gather: dst[j,:] = y[j/10, sub[j]-1, :]
// ---------------------------------------------------------------------------
__global__ void gather_rows(const float* __restrict__ y, const int* __restrict__ sub,
                            float* __restrict__ dst) {
  int j = blockIdx.x, c = threadIdx.x;
  int b = j / 10;
  int t = sub[j] - 1;
  dst[(size_t)j * HDIM + c] = y[((size_t)b * TLEN + t) * HDIM + c];
}

// ---------------------------------------------------------------------------
// FSim on the gathered matrix G (11008 x 128), in place:
//  C = sig(xWi+bi)*tanh(xWc+bc); out = sig(xWo+bo)*tanh(C) + x
// ---------------------------------------------------------------------------
__global__ __launch_bounds__(256) void fsim_kernel(
    float* __restrict__ Gm,
    const _Float16* __restrict__ gWi, const _Float16* __restrict__ gWc,
    const _Float16* __restrict__ gWo,
    const float* __restrict__ bi, const float* __restrict__ bc,
    const float* __restrict__ bo) {
  extern __shared__ char smem[];
  _Float16* sWi = (_Float16*)smem;
  _Float16* sWc = sWi + HDIM * HDIM;
  _Float16* sWo = sWc + HDIM * HDIM;
  float* sBi = (float*)(sWo + HDIM * HDIM);
  float* sBc = sBi + HDIM;
  float* sBo = sBc + HDIM;
  _Float16* sA = (_Float16*)(sBo + HDIM);     // 8 waves * 16 * 128 f16

  int tid = threadIdx.x, lane = tid & 31, wave = tid >> 5;
  for (int i = tid; i < HDIM * HDIM; i += 256) { sWi[i] = gWi[i]; sWc[i] = gWc[i]; sWo[i] = gWo[i]; }
  for (int i = tid; i < HDIM; i += 256) { sBi[i] = bi[i]; sBc[i] = bc[i]; sBo[i] = bo[i]; }

  int rowbase = (blockIdx.x * 8 + wave) * 16;
  _Float16* myA = sA + wave * 16 * HDIM;
  for (int i = lane * 64, e = i + 64; i < e; ++i) {
    int r = i >> 7, c = i & 127;
    myA[i] = (_Float16)Gm[(size_t)(rowbase + r) * HDIM + c];
  }
  __syncthreads();

  int mrBase = (lane & 16) ? 8 : 0;
#pragma unroll 1
  for (int nt = 0; nt < 8; ++nt) {
    int ncol = nt * 16 + (lane & 15);
    v8f ai, ac, ao;
    float b0i = sBi[ncol], b0c = sBc[ncol], b0o = sBo[ncol];
#pragma unroll
    for (int r = 0; r < 8; ++r) { ai[r] = b0i; ac[r] = b0c; ao[r] = b0o; }
#pragma unroll
    for (int kk = 0; kk < 4; ++kk) {
      v16h a = load_a_frag(myA, lane, kk);
      ai = wmma16(a, load_b_frag(sWi, ncol, lane, kk), ai);
      ac = wmma16(a, load_b_frag(sWc, ncol, lane, kk), ac);
      ao = wmma16(a, load_b_frag(sWo, ncol, lane, kk), ao);
    }
#pragma unroll
    for (int r = 0; r < 8; ++r) {
      size_t idx = (size_t)(rowbase + mrBase + r) * HDIM + ncol;
      float xres = Gm[idx];
      float Cv = sigf(ai[r]) * tanhf(ac[r]);
      Gm[idx] = sigf(ao[r]) * tanhf(Cv) + xres;
    }
  }
}

// ---------------------------------------------------------------------------
// One wave32 per output pair: out[p] = exp(-||G[u]-G[v]||)
// ---------------------------------------------------------------------------
__global__ __launch_bounds__(256) void dist_kernel(const float* __restrict__ Gm,
                                                   float* __restrict__ out) {
  int p = (blockIdx.x * blockDim.x + threadIdx.x) >> 5;
  int lane = threadIdx.x & 31;
  int u, v;
  if (p < 256)       { u = p;                 v = 256 + p; }
  else if (p < 512)  { int q = p - 256;  u = q;            v = 512 + q; }
  else if (p < 3072) { int j = p - 512;  u = TA_OFF + j;   v = A_OFF + j; }
  else               { int j = p - 3072; u = TF_OFF + j;   v = F_OFF + j; }
  float s = 0.f;
  for (int c = lane; c < HDIM; c += 32) {
    float d = Gm[(size_t)u * HDIM + c] - Gm[(size_t)v * HDIM + c];
    s += d * d;
  }
#pragma unroll
  for (int off = 16; off > 0; off >>= 1) s += __shfl_xor(s, off, 32);
  if (lane == 0) out[p] = __expf(-sqrtf(s));
}

// ---------------------------------------------------------------------------
extern "C" void kernel_launch(void* const* d_in, const int* in_sizes, int n_in,
                              void* d_out, int out_size, void* d_ws, size_t ws_size,
                              hipStream_t stream) {
  // workspace carve-up (~140 MB)
  char* p = (char*)d_ws;
  float* y = (float*)p;                 p += (size_t)BATCH * TLEN * HDIM * 4;
  float* G = (float*)p;                 p += (size_t)NROWS_G * HDIM * 4;
  _Float16* gWih = (_Float16*)p;        p += (size_t)G4 * HDIM * 2;
  _Float16* gWhh = (_Float16*)p;        p += (size_t)G4 * HDIM * 2;
  float* gWgps = (float*)p;             p += (size_t)G4 * 2 * 4;
  float* gBias = (float*)p;             p += (size_t)G4 * 4;
  _Float16* gWi = (_Float16*)p;         p += (size_t)HDIM * HDIM * 2;
  _Float16* gWc = (_Float16*)p;         p += (size_t)HDIM * HDIM * 2;
  _Float16* gWo = (_Float16*)p;         p += (size_t)HDIM * HDIM * 2;

  const float* E    = (const float*)d_in[19];
  const float* W_ih = (const float*)d_in[20];
  const float* W_hh = (const float*)d_in[21];
  const float* b_ih = (const float*)d_in[22];
  const float* b_hh = (const float*)d_in[23];
  const float* Wi = (const float*)d_in[24];
  const float* bi = (const float*)d_in[25];
  const float* Wc = (const float*)d_in[26];
  const float* bc = (const float*)d_in[27];
  const float* Wo = (const float*)d_in[28];
  const float* bo = (const float*)d_in[29];

  constexpr size_t LSTM_LDS = (size_t)G4 * HDIM * 2 * 2 + G4 * 2 * 4 + G4 * 4 +
                              2 * 16 * HDIM * 2 + 32 * 4 + 16 * 4;      // 276,672 B
  constexpr size_t FSIM_LDS = 3 * (size_t)HDIM * HDIM * 2 + 3 * HDIM * 4 +
                              8 * 16 * HDIM * 2;                        // 132,608 B
  hipFuncSetAttribute((const void*)lstm_kernel,
                      hipFuncAttributeMaxDynamicSharedMemorySize, (int)LSTM_LDS);
  hipFuncSetAttribute((const void*)fsim_kernel,
                      hipFuncAttributeMaxDynamicSharedMemorySize, (int)FSIM_LDS);

  prep_weights<<<256, 256, 0, stream>>>(W_ih, W_hh, b_ih, b_hh, Wi, Wc, Wo,
                                        gWih, gWhh, gWgps, gBias, gWi, gWc, gWo);

  // trajectory inputs: (gps, grid, length) indices per encoder
  const int gpsIdx[3]  = {0, 7, 13};
  const int gridIdx[3] = {2, 9, 15};
  const int lenIdx[3]  = {4, 11, 17};
  for (int traj = 0; traj < 3; ++traj) {
    const float* gps = (const float*)d_in[gpsIdx[traj]];
    const int* grid  = (const int*)d_in[gridIdx[traj]];
    const int* len   = (const int*)d_in[lenIdx[traj]];
    float* hT = G + (size_t)traj * BATCH * HDIM;   // hT rows 0..767 of G
    lstm_kernel<<<16, 256, LSTM_LDS, stream>>>(gps, grid, len, E, gWih, gWhh,
                                               gWgps, gBias, y, hT);
    if (traj == 0) {
      gather_rows<<<2560, 128, 0, stream>>>(y, (const int*)d_in[5], G + (size_t)TA_OFF * HDIM);
      gather_rows<<<2560, 128, 0, stream>>>(y, (const int*)d_in[6], G + (size_t)TF_OFF * HDIM);
    } else if (traj == 1) {
      gather_rows<<<2560, 128, 0, stream>>>(y, (const int*)d_in[12], G + (size_t)A_OFF * HDIM);
    } else {
      gather_rows<<<2560, 128, 0, stream>>>(y, (const int*)d_in[18], G + (size_t)F_OFF * HDIM);
    }
  }

  // FSim over all 11008 gathered/hT rows (86 blocks * 8 waves * 16 rows)
  fsim_kernel<<<NROWS_G / 128, 256, FSIM_LDS, stream>>>(G, gWi, gWc, gWo, bi, bc, bo);

  // 5632 output pairs, one wave each
  dist_kernel<<<5632 / 8, 256, 0, stream>>>(G, (float*)d_out);
}
